// Net_17428977287693
// MI455X (gfx1250) — compile-verified, matrix operand order checked by reading, not run on hardware
//
#include <hip/hip_runtime.h>
#include <math.h>

// ---------------- problem constants ----------------
#define N_NODES   40000
#define N_EDGES   160000
#define N_GRAPHS  1600
#define F_DIM     60
#define T_DIM     5
#define F_OUT     12
#define EDGE_DIM  16
#define TF        300      // T*F
#define K_PRE     192      // 3 x 64 (padded concat of x[dst]|x[src]|eemb)
#define KS_PRE    6        // 192/32
#define TN_PRE    19       // ceil(300/16)
#define K_POST    780      // 13F
#define KS_POST   25       // ceil(780/32)
#define MLPNODE   512
#define MLPOUT    128
#define FINGDIM   2048
#define FINGNODE  512
#define FINGOUT   128
#define PREDNODE  256
#define N_HEADS   7
#define AVG_LOG   1.0450452222917992f

typedef __attribute__((ext_vector_type(16))) __bf16 bf16x16;
typedef __attribute__((ext_vector_type(8)))  __bf16 bf16x8;
typedef __attribute__((ext_vector_type(8)))  float  floatx8;

// fp32 -> bf16 round-to-nearest-even
static __device__ __forceinline__ __bf16 f2bf(float f) {
  unsigned u = __float_as_uint(f);
  u += 0x7FFFu + ((u >> 16) & 1u);
  unsigned short h = (unsigned short)(u >> 16);
  __bf16 o;
  __builtin_memcpy(&o, &h, sizeof(o));
  return o;
}

// order-preserving float<->uint for atomic min/max
static __device__ __forceinline__ unsigned fenc(float f) {
  unsigned u = __float_as_uint(f);
  return (u & 0x80000000u) ? ~u : (u | 0x80000000u);
}
static __device__ __forceinline__ float fdec(unsigned u) {
  unsigned b = (u & 0x80000000u) ? (u & 0x7FFFFFFFu) : ~u;
  return __uint_as_float(b);
}
static __device__ __forceinline__ float finite0(float v) {
  unsigned u = __float_as_uint(v);
  return ((u & 0x7F800000u) == 0x7F800000u) ? 0.f : v;
}

// CDNA5 async global->LDS copy of 16 bytes (GV addressing), tracked by ASYNCcnt
static __device__ __forceinline__ void async_copy_b128(unsigned lds_addr, const void* gaddr) {
  asm volatile("global_load_async_to_lds_b128 %0, %1, off"
               :: "v"(lds_addr), "v"(gaddr) : "memory");
}
static __device__ __forceinline__ void async_wait0() {
  asm volatile("s_wait_asynccnt 0" ::: "memory");
}

// per-lane A fragment from an LDS 16x32 f32 slab (row-major, stride 32)
static __device__ __forceinline__ bf16x16 frag_from_lds_f32(const float* rawA, int lane,
                                                            int kbase, int K) {
  const int row = lane & 15, half = lane >> 4;
  float av[16];
  const float4* rp  = (const float4*)&rawA[row * 32 + half * 8];
  const float4* rp2 = (const float4*)&rawA[row * 32 + 16 + half * 8];
  ((float4*)av)[0] = rp[0];
  ((float4*)av)[1] = rp[1];
  ((float4*)av)[2] = rp2[0];
  ((float4*)av)[3] = rp2[1];
  bf16x16 a;
#pragma unroll
  for (int e = 0; e < 16; ++e) {
    int kofs = ((e >= 8) ? 16 : 0) + half * 8 + (e & 7);
    float v = (kbase + kofs < K) ? av[e] : 0.f;
    a[e] = f2bf(v);
  }
  return a;
}

// ---------------- B-matrix fragment packer ----------------
// out flat layout: ((tn*KS + ks)*32 + lane)*16 + e   (bf16)
__global__ void pack_b_kernel(const float* __restrict__ W, __bf16* __restrict__ out,
                              int K, int N) {
  const int KS = (K + 31) >> 5, TN = (N + 15) >> 4;
  const long total = (long)TN * KS * 512;
  long i = (long)blockIdx.x * blockDim.x + threadIdx.x;
  if (i >= total) return;
  int e    = (int)(i & 15);
  int lane = (int)((i >> 4) & 31);
  long rest = i >> 9;
  int ks = (int)(rest % KS);
  int tn = (int)(rest / KS);
  int half = lane >> 4, lr = lane & 15;
  int kofs = ((e >= 8) ? 16 : 0) + half * 8 + (e & 7);
  int k = ks * 32 + kofs, col = tn * 16 + lr;
  float v = (k < K && col < N) ? W[(long)k * N + col] : 0.f;
  out[i] = f2bf(v);
}

// pack pre_W [T,180,60] into fragments over padded K=192 (3 sources x 64),
// cols c = t*60+g (300 total). logical weight row = part*60 + (k&63), zero if (k&63)>=60.
__global__ void pack_pre_kernel(const float* __restrict__ preW, __bf16* __restrict__ out) {
  const long total = (long)TN_PRE * KS_PRE * 512;
  long i = (long)blockIdx.x * blockDim.x + threadIdx.x;
  if (i >= total) return;
  int e    = (int)(i & 15);
  int lane = (int)((i >> 4) & 31);
  long rest = i >> 9;
  int ks = (int)(rest % KS_PRE);
  int tn = (int)(rest / KS_PRE);
  int half = lane >> 4, lr = lane & 15;
  int kofs = ((e >= 8) ? 16 : 0) + half * 8 + (e & 7);
  int k = ks * 32 + kofs;              // 0..191
  int part = k >> 6, kk = k & 63;      // source, in-source row
  int c = tn * 16 + lr;
  float v = 0.f;
  if (kk < 60 && c < TF) {
    int t = c / F_DIM, g = c - t * F_DIM;
    v = preW[((long)t * 180 + part * 60 + kk) * F_DIM + g];
  }
  out[i] = f2bf(v);
}

// convert [M,60] f32 -> [M,64] bf16, zero padded
__global__ void cvt_pad64_kernel(const float* __restrict__ X, __bf16* __restrict__ Y, int M) {
  long i = (long)blockIdx.x * blockDim.x + threadIdx.x;
  if (i >= (long)M * 64) return;
  int c = (int)(i & 63);
  long r = i >> 6;
  Y[i] = (c < F_DIM) ? f2bf(X[r * F_DIM + c]) : f2bf(0.f);
}

// ---------------- generic WMMA GEMM with async-LDS A staging -----------------
// C = act(A[M,K] * W + bias); W prepacked bf16 fragments. Block: 128 thr = 4 waves,
// covers 16 rows x (4 x 16) cols sharing one staged A slab. Requires K % 4 == 0.
template <int ACT>
__global__ void gemm_lds_kernel(const float* __restrict__ A, const __bf16* __restrict__ Bp,
                                const float* __restrict__ bias, float* __restrict__ C,
                                int M, int N, int K) {
  __shared__ __align__(16) float rawA[16 * 32];
  const int tid = threadIdx.x, lane = tid & 31, wave = tid >> 5;
  const int tilesN = (N + 15) >> 4;
  const int tn4 = (tilesN + 3) >> 2;
  const int bm = blockIdx.x / tn4, bn4 = blockIdx.x % tn4;
  const int row0 = bm << 4;
  const int tn = bn4 * 4 + wave;
  const bool active = (tn < tilesN);            // wave-uniform
  const int KS = (K + 31) >> 5;
  const unsigned ldsA = (unsigned)(size_t)&rawA[0];
  const int rrow = tid >> 3, seg = tid & 7;     // staging role: 16 rows x 8 segs
  floatx8 acc = {};
  for (int ks = 0; ks < KS; ++ks) {
    int kseg = ks * 32 + seg * 4;
    if (kseg + 3 < K)
      async_copy_b128(ldsA + (unsigned)(rrow * 32 + seg * 4) * 4u,
                      &A[(long)(row0 + rrow) * K + kseg]);
    async_wait0();
    __syncthreads();
    if (active) {
      bf16x16 a = frag_from_lds_f32(rawA, lane, ks * 32, K);
      bf16x16 b = ((const bf16x16*)Bp)[(tn * KS + ks) * 32 + lane];
      acc = __builtin_amdgcn_wmma_f32_16x16x32_bf16(false, a, false, b, (short)0, acc,
                                                    false, false);
    }
    __syncthreads();
  }
  if (active) {
    const int half = lane >> 4, lr = lane & 15;
#pragma unroll
    for (int i = 0; i < 8; ++i) {
      int r = row0 + i + 8 * half, c = (tn << 4) + lr;
      if (r < M && c < N) {
        float v = acc[i] + bias[c];
        if (ACT == 1) v = fmaxf(v, 0.f);
        C[(long)r * N + c] = v;
      }
    }
  }
}

// ---------------- fused PNA pre-einsum + segment scatter (all-bf16 hot loop) --------
// A rows: padded bf16 [xh[dst](64) | xh[src](64) | eh[e](64)], K=192.
// m = A @ packed preW + preB ; scatter to s1/s2/min/max by dst.
__global__ void pna_pre_scatter_kernel(const __bf16* __restrict__ xh,   // [N,64] bf16
                                       const __bf16* __restrict__ eh,   // [E,64] bf16
                                       const __bf16* __restrict__ Bp,   // packed [19][6][512]
                                       const float* __restrict__ preB,  // [300]
                                       const int*  __restrict__ ei,     // [2,E]
                                       float* __restrict__ s1, float* __restrict__ s2,
                                       unsigned* __restrict__ mnU, unsigned* __restrict__ mxU) {
  __shared__ __align__(16) __bf16 rawA[16 * 32];  // 16 rows x 32 bf16 per k-step
  __shared__ int sds[32];                          // [0..15]=dst, [16..31]=src
  const int tid = threadIdx.x, lane = tid & 31, wave = tid >> 5;
  const int bm = blockIdx.x / 5, bn4 = blockIdx.x % 5;  // 5 col groups cover 19 tiles
  const int row0 = bm << 4;
  const int tn = bn4 * 4 + wave;
  const bool active = (tn < TN_PRE);            // wave-uniform
  if (tid < 16) {
    sds[tid]      = ei[N_EDGES + row0 + tid];   // dst
    sds[16 + tid] = ei[row0 + tid];             // src
  }
  __syncthreads();
  const unsigned ldsA = (unsigned)(size_t)&rawA[0];
  const int rrow = tid >> 2, seg = tid & 3;     // staging: 16 rows x 4 segs (8 bf16 = 16B)
  floatx8 acc = {};
  for (int ks = 0; ks < KS_PRE; ++ks) {
    if (tid < 64) {                             // waves 0-1 stage; uniform per wave
      int k8 = ks * 32 + seg * 8;               // 0..184, multiple of 8
      int part = k8 >> 6, ofs = k8 & 63;
      const __bf16* g;
      if (part == 0)      g = &xh[(long)sds[rrow] * 64 + ofs];
      else if (part == 1) g = &xh[(long)sds[16 + rrow] * 64 + ofs];
      else                g = &eh[(long)(row0 + rrow) * 64 + ofs];
      async_copy_b128(ldsA + (unsigned)(rrow * 32 + seg * 8) * 2u, g);
    }
    async_wait0();
    __syncthreads();
    if (active) {
      const int row = lane & 15, half = lane >> 4;
      bf16x8 alo = *(const bf16x8*)&rawA[row * 32 + half * 8];
      bf16x8 ahi = *(const bf16x8*)&rawA[row * 32 + 16 + half * 8];
      bf16x16 a;
#pragma unroll
      for (int e = 0; e < 8; ++e) { a[e] = alo[e]; a[8 + e] = ahi[e]; }
      bf16x16 b = ((const bf16x16*)Bp)[(tn * KS_PRE + ks) * 32 + lane];
      acc = __builtin_amdgcn_wmma_f32_16x16x32_bf16(false, a, false, b, (short)0, acc,
                                                    false, false);
    }
    __syncthreads();
  }
  if (active) {
    const int half = lane >> 4, lr = lane & 15;
#pragma unroll
    for (int i = 0; i < 8; ++i) {
      int c = (tn << 4) + lr;
      if (c < TF) {
        float v = acc[i] + preB[c];
        int d = sds[i + 8 * half];
        long idx = (long)d * TF + c;
        atomicAdd(&s1[idx], v);
        atomicAdd(&s2[idx], v * v);
        unsigned u = fenc(v);
        atomicMax(&mxU[idx], u);
        atomicMin(&mnU[idx], u);
      }
    }
  }
}

// ---------------- degree / scales / aggregate finalize ----------------
__global__ void degree_kernel(const int* __restrict__ ei, float* __restrict__ deg) {
  int e = blockIdx.x * blockDim.x + threadIdx.x;
  if (e >= N_EDGES) return;
  atomicAdd(&deg[ei[N_EDGES + e]], 1.f);
}

__global__ void finalize_agg_kernel(float* __restrict__ s1, float* __restrict__ s2,
                                    unsigned* __restrict__ mnU, unsigned* __restrict__ mxU,
                                    const float* __restrict__ deg) {
  long i = (long)blockIdx.x * blockDim.x + threadIdx.x;
  if (i >= (long)N_NODES * TF) return;
  int n = (int)(i / TF);
  float degc = fmaxf(deg[n], 1.f);
  float mean = s1[i] / degc;
  s1[i] = mean;
  float var = s2[i] / degc - mean * mean;
  s2[i] = sqrtf(fmaxf(var, 0.f) + 1e-5f);          // std
  ((float*)mnU)[i] = finite0(fdec(mnU[i]));
  ((float*)mxU)[i] = finite0(fdec(mxU[i]));
}

__global__ void node_scales_kernel(const float* __restrict__ deg,
                                   float* __restrict__ scl1, float* __restrict__ scl2) {
  int n = blockIdx.x * blockDim.x + threadIdx.x;
  if (n >= N_NODES) return;
  float degc = fmaxf(deg[n], 1.f);
  float logd = logf(degc + 1.f);
  scl1[n] = logd / AVG_LOG;
  scl2[n] = AVG_LOG / logd;
}

// ---------------- PNA post einsum (register gather A, packed B) ----------------
__global__ void pna_post_kernel(const float* __restrict__ x,
                                const float* __restrict__ meanB, const float* __restrict__ stdB,
                                const float* __restrict__ mnB,   const float* __restrict__ mxB,
                                const float* __restrict__ scl1,  const float* __restrict__ scl2,
                                const __bf16* __restrict__ Bp,   // packed, 12800 per t
                                const float* __restrict__ postB, // [5,12]
                                float* __restrict__ o) {         // [N,60]
  const int lane = threadIdx.x & 31;
  const int wave = threadIdx.x >> 5;
  const int wpb  = blockDim.x >> 5;
  const int tilesM = N_NODES / 16;
  const int tile = blockIdx.x * wpb + wave;
  if (tile >= tilesM) return;        // wave-uniform
  const int row0 = tile << 4;
  const int half = lane >> 4, lr = lane & 15;
  const int r = row0 + lr;
  const float sA1 = scl1[r], sA2 = scl2[r];
  for (int t = 0; t < T_DIM; ++t) {
    floatx8 acc = {};
    for (int ks = 0; ks < KS_POST; ++ks) {
      bf16x16 a, b;
#pragma unroll
      for (int e = 0; e < 16; ++e) {
        int k = ks * 32 + ((e >= 8) ? 16 : 0) + half * 8 + (e & 7);
        float av = 0.f;
        if (k < 60) {
          av = x[(long)r * F_DIM + k];
        } else if (k < K_POST) {
          int kk = k - 60;
          int grp = kk / 240;
          int idx = kk - grp * 240;
          int part = idx / 60;
          int f = idx - part * 60;
          long ci = (long)r * TF + t * F_DIM + f;
          float v = (part == 0) ? meanB[ci] : (part == 1) ? mnB[ci]
                  : (part == 2) ? mxB[ci]   : stdB[ci];
          if (grp == 1) v *= sA1;
          else if (grp == 2) v *= sA2;
          av = v;
        }
        a[e] = f2bf(av);
      }
      b = ((const bf16x16*)Bp)[(t * KS_POST + ks) * 32 + lane];
      acc = __builtin_amdgcn_wmma_f32_16x16x32_bf16(false, a, false, b, (short)0, acc,
                                                    false, false);
    }
#pragma unroll
    for (int i = 0; i < 8; ++i) {
      int rr = row0 + i + 8 * half;
      if (lr < F_OUT)
        o[(long)rr * (T_DIM * F_OUT) + t * F_OUT + lr] = acc[i] + postB[t * F_OUT + lr];
    }
  }
}

// ---------------- batchnorm (biased variance, torch-style) ----------------
__global__ void bn_stats_kernel(const float* __restrict__ X, float* __restrict__ sums,
                                float* __restrict__ sqs, int M, int N) {
  int c = blockIdx.x * blockDim.x + threadIdx.x;
  if (c >= N) return;
  float s = 0.f, q = 0.f;
  for (int r = 0; r < M; ++r) {
    float v = X[(long)r * N + c];
    s += v; q += v * v;
  }
  sums[c] = s; sqs[c] = q;
}

template <int RELU>
__global__ void bn_apply_kernel(const float* __restrict__ X, float* __restrict__ Y,
                                const float* __restrict__ sums, const float* __restrict__ sqs,
                                const float* __restrict__ gamma, const float* __restrict__ beta,
                                int M, int N) {
  long i = (long)blockIdx.x * blockDim.x + threadIdx.x;
  if (i >= (long)M * N) return;
  int c = (int)(i % N);
  float mean = sums[c] / (float)M;
  float var  = sqs[c] / (float)M - mean * mean;
  float v = gamma[c] * (X[i] - mean) * rsqrtf(var + 1e-5f) + beta[c];
  if (RELU) v = fmaxf(v, 0.f);
  Y[i] = v;
}

// ---------------- misc ----------------
__global__ void pool_kernel(const float* __restrict__ h, const int* __restrict__ batch,
                            float* __restrict__ pooled) {
  long i = (long)blockIdx.x * blockDim.x + threadIdx.x;
  if (i >= (long)N_NODES * F_DIM) return;
  int n = (int)(i / F_DIM);
  int f = (int)(i - (long)n * F_DIM);
  atomicAdd(&pooled[(long)batch[n] * F_DIM + f], h[i]);
}

__global__ void concat_kernel(const float* __restrict__ a, const float* __restrict__ b,
                              float* __restrict__ out, int M, int Na, int Nb) {
  long i = (long)blockIdx.x * blockDim.x + threadIdx.x;
  int Nt = Na + Nb;
  if (i >= (long)M * Nt) return;
  int r = (int)(i / Nt), c = (int)(i - (long)r * Nt);
  out[i] = (c < Na) ? a[(long)r * Na + c] : b[(long)r * Nb + (c - Na)];
}

__global__ void fillu_kernel(unsigned* __restrict__ p, unsigned v, long n) {
  long i = (long)blockIdx.x * blockDim.x + threadIdx.x;
  if (i < n) p[i] = v;
}

static inline int ceil_div_h(long a, long b) { return (int)((a + b - 1) / b); }

// ---------------- host orchestration ----------------
extern "C" void kernel_launch(void* const* d_in, const int* in_sizes, int n_in,
                              void* d_out, int out_size, void* d_ws, size_t ws_size,
                              hipStream_t stream) {
  (void)in_sizes; (void)n_in; (void)out_size; (void)ws_size;

  // ---- inputs, flattened in setup_inputs() insertion order ----
  const float* x_in      = (const float*)d_in[0];
  const float* edge_attr = (const float*)d_in[1];
  const float* fing      = (const float*)d_in[2];
  // pna layer l: base 3 + l*10 : edge_W, edge_b, pre_W, pre_b, post_W, post_b, lin_W, lin_b, bn_g, bn_b
  const float* mlpW1 = (const float*)d_in[23]; const float* mlpb1 = (const float*)d_in[24];
  const float* mlpW2 = (const float*)d_in[25]; const float* mlpb2 = (const float*)d_in[26];
  const float* mlpW3 = (const float*)d_in[27]; const float* mlpb3 = (const float*)d_in[28];
  const float* fW1 = (const float*)d_in[29]; const float* fb1 = (const float*)d_in[30];
  const float* fg1 = (const float*)d_in[31]; const float* fe1 = (const float*)d_in[32];
  const float* fW2 = (const float*)d_in[33]; const float* fb2 = (const float*)d_in[34];
  const float* fg2 = (const float*)d_in[35]; const float* fe2 = (const float*)d_in[36];
  const float* fW3 = (const float*)d_in[37]; const float* fb3 = (const float*)d_in[38];
  const float* connW = (const float*)d_in[39]; const float* connb = (const float*)d_in[40];
  const float* headW = (const float*)d_in[41]; const float* headb = (const float*)d_in[42];
  const int* ei    = (const int*)d_in[43];   // [2,E]
  const int* batch = (const int*)d_in[44];   // [N]
  float* out = (float*)d_out;                // [1600,7]

  // ---- workspace carve-out ----
  char* p = (char*)d_ws;
  auto takeB = [&](long bytes) {
    void* r = (void*)p;
    p += ((bytes + 255) / 256) * 256;
    return r;
  };
  auto takeF = [&](long elems) { return takeB(elems * 4); };
  auto takeH = [&](long elems) { return takeB(elems * 2); };  // bf16

  float*    eemb  = (float*)   takeF((long)N_EDGES * F_DIM);
  float*    s1    = (float*)   takeF((long)N_NODES * TF);
  float*    s2    = (float*)   takeF((long)N_NODES * TF);
  unsigned* mnU   = (unsigned*)takeF((long)N_NODES * TF);
  unsigned* mxU   = (unsigned*)takeF((long)N_NODES * TF);
  float*    deg   = (float*)   takeF(N_NODES);
  float*    scl1  = (float*)   takeF(N_NODES);
  float*    scl2  = (float*)   takeF(N_NODES);
  float*    obuf  = (float*)   takeF((long)N_NODES * F_DIM);
  float*    tmp   = (float*)   takeF((long)N_NODES * F_DIM);
  float*    hbuf  = (float*)   takeF((long)N_NODES * F_DIM);
  float*    sums  = (float*)   takeF(2048);
  float*    sqs   = (float*)   takeF(2048);
  float*    pooled= (float*)   takeF((long)N_GRAPHS * F_DIM);
  float*    g1    = (float*)   takeF((long)N_GRAPHS * MLPNODE);
  float*    g2    = (float*)   takeF((long)N_GRAPHS * MLPNODE);
  float*    g3    = (float*)   takeF((long)N_GRAPHS * MLPOUT);
  float*    f1    = (float*)   takeF((long)N_GRAPHS * FINGNODE);
  float*    f2    = (float*)   takeF((long)N_GRAPHS * FINGNODE);
  float*    f3    = (float*)   takeF((long)N_GRAPHS * FINGOUT);
  float*    dmr   = (float*)   takeF((long)N_GRAPHS * (MLPOUT + FINGOUT));
  float*    dmr2  = (float*)   takeF((long)N_GRAPHS * PREDNODE);
  __bf16*   xh    = (__bf16*)  takeH((long)N_NODES * 64);      // padded bf16 node feats
  __bf16*   eh    = (__bf16*)  takeH((long)N_EDGES * 64);      // padded bf16 edge embeds
  __bf16*   pkEdge= (__bf16*)  takeH(4L * 1 * 512);            // K=16,N=60
  __bf16*   pkPre = (__bf16*)  takeH((long)TN_PRE * KS_PRE * 512);
  __bf16*   pkPost= (__bf16*)  takeH(5L * KS_POST * 512);
  __bf16*   pkLin = (__bf16*)  takeH(4L * 2 * 512);            // K=60,N=60
  __bf16*   pkM1  = (__bf16*)  takeH(32L * 2 * 512);
  __bf16*   pkM2  = (__bf16*)  takeH(32L * 16 * 512);
  __bf16*   pkM3  = (__bf16*)  takeH(8L * 16 * 512);
  __bf16*   pkF1  = (__bf16*)  takeH(32L * 64 * 512);
  __bf16*   pkF2  = (__bf16*)  takeH(32L * 16 * 512);
  __bf16*   pkF3  = (__bf16*)  takeH(8L * 16 * 512);
  __bf16*   pkCn  = (__bf16*)  takeH(16L * 8 * 512);
  __bf16*   pkHd  = (__bf16*)  takeH(1L * 8 * 512);

  auto pack = [&](const float* W, __bf16* o, int K, int N) {
    long total = (long)((N + 15) / 16) * ((K + 31) / 32) * 512;
    pack_b_kernel<<<ceil_div_h(total, 256), 256, 0, stream>>>(W, o, K, N);
  };
  auto gemm = [&](const float* A, const __bf16* Bp, const float* bias, float* C,
                  int M, int N, int K, int relu) {
    int tilesM = (M + 15) / 16, tn4 = ((N + 15) / 16 + 3) / 4;
    int blocks = tilesM * tn4;
    if (relu) gemm_lds_kernel<1><<<blocks, 128, 0, stream>>>(A, Bp, bias, C, M, N, K);
    else      gemm_lds_kernel<0><<<blocks, 128, 0, stream>>>(A, Bp, bias, C, M, N, K);
  };

  // ---- degree + scalers (dst fixed across layers) ----
  hipMemsetAsync(deg, 0, N_NODES * sizeof(float), stream);
  degree_kernel<<<ceil_div_h(N_EDGES, 256), 256, 0, stream>>>(ei, deg);
  node_scales_kernel<<<ceil_div_h(N_NODES, 256), 256, 0, stream>>>(deg, scl1, scl2);

  // ---- pack static weights once ----
  pack(mlpW1, pkM1, F_DIM, MLPNODE);
  pack(mlpW2, pkM2, MLPNODE, MLPNODE);
  pack(mlpW3, pkM3, MLPNODE, MLPOUT);
  pack(fW1,   pkF1, FINGDIM, FINGNODE);
  pack(fW2,   pkF2, FINGNODE, FINGNODE);
  pack(fW3,   pkF3, FINGNODE, FINGOUT);
  pack(connW, pkCn, MLPOUT + FINGOUT, PREDNODE);
  pack(headW, pkHd, PREDNODE, N_HEADS);

  const long NA = (long)N_NODES * TF;
  const float* hin = x_in;
  for (int l = 0; l < 2; ++l) {
    const float* edge_W = (const float*)d_in[3 + l * 10 + 0];
    const float* edge_b = (const float*)d_in[3 + l * 10 + 1];
    const float* pre_W  = (const float*)d_in[3 + l * 10 + 2];
    const float* pre_b  = (const float*)d_in[3 + l * 10 + 3];
    const float* post_W = (const float*)d_in[3 + l * 10 + 4];
    const float* post_b = (const float*)d_in[3 + l * 10 + 5];
    const float* lin_W  = (const float*)d_in[3 + l * 10 + 6];
    const float* lin_b  = (const float*)d_in[3 + l * 10 + 7];
    const float* bn_g   = (const float*)d_in[3 + l * 10 + 8];
    const float* bn_b   = (const float*)d_in[3 + l * 10 + 9];

    // pack this layer's weights
    pack(edge_W, pkEdge, EDGE_DIM, F_DIM);
    {
      long total = (long)TN_PRE * KS_PRE * 512;
      pack_pre_kernel<<<ceil_div_h(total, 256), 256, 0, stream>>>(pre_W, pkPre);
    }
    for (int t = 0; t < T_DIM; ++t)
      pack(post_W + (long)t * K_POST * F_OUT, pkPost + (long)t * KS_POST * 512, K_POST, F_OUT);
    pack(lin_W, pkLin, T_DIM * F_OUT, F_DIM);

    // edge embedding: eemb = edge_attr @ edge_W + edge_b   [E,60]
    gemm(edge_attr, pkEdge, edge_b, eemb, N_EDGES, F_DIM, EDGE_DIM, 0);

    // bf16 padded copies of activations for the edge GEMM
    cvt_pad64_kernel<<<ceil_div_h((long)N_NODES * 64, 256), 256, 0, stream>>>(hin, xh, N_NODES);
    cvt_pad64_kernel<<<ceil_div_h((long)N_EDGES * 64, 256), 256, 0, stream>>>(eemb, eh, N_EDGES);

    // init aggregation buffers
    hipMemsetAsync(s1, 0, NA * sizeof(float), stream);
    hipMemsetAsync(s2, 0, NA * sizeof(float), stream);
    fillu_kernel<<<ceil_div_h(NA, 256), 256, 0, stream>>>(mnU, 0xFF800000u, NA); // enc(+inf)
    fillu_kernel<<<ceil_div_h(NA, 256), 256, 0, stream>>>(mxU, 0x007FFFFFu, NA); // enc(-inf)

    // fused pre-einsum + segment scatter (10000 edge tiles x 5 col groups)
    pna_pre_scatter_kernel<<<(N_EDGES / 16) * 5, 128, 0, stream>>>(
        xh, eh, pkPre, pre_b, ei, s1, s2, mnU, mxU);

    // mean/std/min/max finalize
    finalize_agg_kernel<<<ceil_div_h(NA, 256), 256, 0, stream>>>(s1, s2, mnU, mxU, deg);

    // post einsum -> o [N,60]
    pna_post_kernel<<<ceil_div_h(N_NODES / 16, 4), 128, 0, stream>>>(
        hin, s1, s2, (const float*)mnU, (const float*)mxU, scl1, scl2, pkPost, post_b, obuf);

    // lin -> tmp [N,60]
    gemm(obuf, pkLin, lin_b, tmp, N_NODES, F_DIM, T_DIM * F_OUT, 0);

    // batchnorm + relu -> hbuf
    bn_stats_kernel<<<1, 64, 0, stream>>>(tmp, sums, sqs, N_NODES, F_DIM);
    bn_apply_kernel<1><<<ceil_div_h((long)N_NODES * F_DIM, 256), 256, 0, stream>>>(
        tmp, hbuf, sums, sqs, bn_g, bn_b, N_NODES, F_DIM);
    hin = hbuf;
  }

  // ---- global add pool ----
  hipMemsetAsync(pooled, 0, (long)N_GRAPHS * F_DIM * sizeof(float), stream);
  pool_kernel<<<ceil_div_h((long)N_NODES * F_DIM, 256), 256, 0, stream>>>(hbuf, batch, pooled);

  // ---- graph MLP branch ----
  gemm(pooled, pkM1, mlpb1, g1, N_GRAPHS, MLPNODE, F_DIM, 1);
  gemm(g1,     pkM2, mlpb2, g2, N_GRAPHS, MLPNODE, MLPNODE, 1);
  gemm(g2,     pkM3, mlpb3, g3, N_GRAPHS, MLPOUT,  MLPNODE, 0);

  // ---- fingerprint branch (Linear -> BN -> ReLU, twice; then Linear) ----
  gemm(fing, pkF1, fb1, f1, N_GRAPHS, FINGNODE, FINGDIM, 0);
  bn_stats_kernel<<<ceil_div_h(FINGNODE, 64), 64, 0, stream>>>(f1, sums, sqs, N_GRAPHS, FINGNODE);
  bn_apply_kernel<1><<<ceil_div_h((long)N_GRAPHS * FINGNODE, 256), 256, 0, stream>>>(
      f1, f1, sums, sqs, fg1, fe1, N_GRAPHS, FINGNODE);
  gemm(f1, pkF2, fb2, f2, N_GRAPHS, FINGNODE, FINGNODE, 0);
  bn_stats_kernel<<<ceil_div_h(FINGNODE, 64), 64, 0, stream>>>(f2, sums, sqs, N_GRAPHS, FINGNODE);
  bn_apply_kernel<1><<<ceil_div_h((long)N_GRAPHS * FINGNODE, 256), 256, 0, stream>>>(
      f2, f2, sums, sqs, fg2, fe2, N_GRAPHS, FINGNODE);
  gemm(f2, pkF3, fb3, f3, N_GRAPHS, FINGOUT, FINGNODE, 0);

  // ---- concat + prediction head ----
  concat_kernel<<<ceil_div_h((long)N_GRAPHS * (MLPOUT + FINGOUT), 256), 256, 0, stream>>>(
      g3, f3, dmr, N_GRAPHS, MLPOUT, FINGOUT);
  gemm(dmr,  pkCn, connb, dmr2, N_GRAPHS, PREDNODE, MLPOUT + FINGOUT, 1);
  gemm(dmr2, pkHd, headb, out,  N_GRAPHS, N_HEADS,  PREDNODE, 0);
}
